// CricketPredictor_59064390254703
// MI455X (gfx1250) — matrix operand order repaired and checked by reading, not compiled, not run on hardware
//
#include <hip/hip_runtime.h>

typedef _Float16 f16;
typedef __attribute__((ext_vector_type(16))) _Float16 v16h;
typedef __attribute__((ext_vector_type(8)))  _Float16 v8h;
typedef __attribute__((ext_vector_type(8)))  float    v8f;

#define B_TOTAL 16384

// ---- swizzled weight arena offsets (in f16 elements) ----
#define GATW_BASE    0        // 4 x 128x128
#define TOKINIT_BASE 65536    // 160x128
#define WQKV_BASE    86016    // 128x384
#define WO_BASE      135168   // 128x128
#define W1_BASE      151552   // 128x512
#define W2_BASE      217088   // 512x128
#define TL_STRIDE    196608
#define FUSE1_BASE   479232   // 256x128
#define FUSE2_BASE   512000   // 128x64
#define OUTW_BASE    520192   // 64x16 (7 real cols)
#define WS_HALVES    521216

struct P {
  const float *fuse_b1,*fuse_b2,*fuse_w1,*fuse_w2;
  const float *gat_a_dst[4],*gat_a_src[4],*gat_w[4];
  const float *out_b,*out_w,*player_emb,*pos;
  const float *proj_b[17],*proj_w[17];
  const float *t_bat_w,*t_bowl_w,*t_feat_b,*t_feat_w,*team_emb;
  const float *tl_b1[2],*tl_b2[2],*tl_ln1b[2],*tl_ln1s[2],*tl_ln2b[2],*tl_ln2s[2],
              *tl_w1[2],*tl_w2[2],*tl_wo[2],*tl_wqkv[2];
  const float *venue_emb;
  const int *batter_idx,*bowler_idx,*venue_idx,*batting_team_idx,*bowling_team_idx;
  const float *state,*chase,*hr,*hw,*ho;
  const int *hb,*hbl;
  f16 *wsW;
  float *gat_out,*temporal_out;
};

// node order (graph) -> sorted-proj-leaf index, feature offsets/dims
__device__ const int PROJ_IDX[17] = {15,1,5,11,6,9,14,16,12,13,2,3,8,0,4,10,7};
__device__ const int FOFF[17]     = {0,32,64,96,100,103,107,110,112,176,182,246,252,256,257,258,259};
__device__ const int FDIM[17]     = {32,32,32,4,3,4,3,2,64,6,64,6,4,1,1,1,2};

// -------------------- WMMA fragment helpers --------------------
// A fragment (16x32 f16, ISA layout): lane = half*16+m reads K = half*8..+7 and 16+half*8..+7
__device__ __forceinline__ v16h load_a(const f16* A, int lda, int lane, int rt, int kt) {
  const int m  = rt*16 + (lane & 15);
  const int c0 = kt*32 + ((lane >> 4) << 3);
  const f16* q = A + m*lda + c0;
  v16h a;
  *(v8h*)&a       = *(const v8h*)q;
  *((v8h*)&a + 1) = *(const v8h*)(q + 16);
  return a;
}
// B fragment: pre-swizzled, 32 contiguous bytes per lane
__device__ __forceinline__ v16h load_b(const f16* tile, int lane) {
  return *(const v16h*)(tile + lane*16);
}

template <int KT, int NT, typename Epi>
__device__ __forceinline__ void gemm_a_res(const f16* A, int lda, const f16* Bw,
                                           int lane, int rt, Epi epi) {
  v16h a[KT];
#pragma unroll
  for (int kt = 0; kt < KT; ++kt) a[kt] = load_a(A, lda, lane, rt, kt);
#pragma unroll
  for (int nt = 0; nt < NT; ++nt) {
    v8f acc = {};
#pragma unroll
    for (int kt = 0; kt < KT; ++kt)
      acc = __builtin_amdgcn_wmma_f32_16x16x32_f16(false, a[kt], false,
              load_b(Bw + (nt*KT + kt)*512, lane), (short)0, acc, false, false);
    epi(nt, acc);
  }
}

template <int KT, int NT, typename Epi>
__device__ __forceinline__ void gemm_c_res(const f16* A, int lda, const f16* Bw,
                                           int lane, int rt, Epi epi) {
  v8f acc[NT];
  v8f z = {};
#pragma unroll
  for (int nt = 0; nt < NT; ++nt) acc[nt] = z;
#pragma unroll
  for (int kt = 0; kt < KT; ++kt) {
    v16h a = load_a(A, lda, lane, rt, kt);
#pragma unroll
    for (int nt = 0; nt < NT; ++nt)
      acc[nt] = __builtin_amdgcn_wmma_f32_16x16x32_f16(false, a, false,
                  load_b(Bw + (nt*KT + kt)*512, lane), (short)0, acc[nt], false, false);
  }
#pragma unroll
  for (int nt = 0; nt < NT; ++nt) epi(nt, acc[nt]);
}

// -------------------- weight prep / swizzle --------------------
__device__ const int JOB_TILES[16] = {32,32,32,32,40,96,32,128,128,96,32,128,128,64,16,2};
__device__ const int JOB_K[16]     = {128,128,128,128,160,128,128,128,512,128,128,128,512,256,128,64};
__device__ const int JOB_NREAL[16] = {128,128,128,128,128,384,128,512,128,384,128,512,128,128,64,7};
__device__ const int JOB_BASE[16]  = {0,16384,32768,49152,65536,86016,135168,151552,217088,
                                      282624,331776,348160,413696,479232,512000,520192};

__device__ float srcval(const P& p, int job, int k, int n) {
  switch (job) {
    case 0: case 1: case 2: case 3: return p.gat_w[job][k*128 + n];
    case 4:
      if (k < 3)   return p.t_feat_w[k*128 + n];
      if (k < 67)  return p.t_bat_w[(k-3)*128 + n];
      if (k < 131) return p.t_bowl_w[(k-67)*128 + n];
      return 0.f;
    case 5:  return p.tl_wqkv[0][k*384 + n];
    case 6:  return p.tl_wo[0][k*128 + n];
    case 7:  return p.tl_w1[0][k*512 + n];
    case 8:  return p.tl_w2[0][k*128 + n];
    case 9:  return p.tl_wqkv[1][k*384 + n];
    case 10: return p.tl_wo[1][k*128 + n];
    case 11: return p.tl_w1[1][k*512 + n];
    case 12: return p.tl_w2[1][k*128 + n];
    case 13: return p.fuse_w1[k*128 + n];
    case 14: return p.fuse_w2[k*64 + n];
    default: return p.out_w[k*7 + n];
  }
}

__global__ void __launch_bounds__(32) k_prep(P p) {
  int t = blockIdx.x;
  int job = 0;
  while (t >= JOB_TILES[job]) { t -= JOB_TILES[job]; ++job; }
  const int Kt = JOB_K[job] >> 5;
  const int nt = t / Kt, kt = t - nt*Kt;
  const int lane = threadIdx.x;
  f16* dst = p.wsW + JOB_BASE[job] + (nt*Kt + kt)*512 + lane*16;
  const int n  = nt*16 + (lane & 15);
  const int kb = kt*32 + ((lane >> 4) << 3);
#pragma unroll
  for (int j = 0; j < 16; ++j) {
    int k = kb + j + ((j >> 3) << 3);
    float v = 0.f;
    if (n < JOB_NREAL[job] && k < JOB_K[job]) v = srcval(p, job, k, n);
    dst[j] = (f16)v;
  }
}

// -------------------- kernel 1: features + proj + GAT --------------------
#define K1_WAVE 18048
__global__ void __launch_bounds__(128) k_gat(P p) {
  extern __shared__ char smem[];
  const int lane = threadIdx.x & 31;
  const int wid  = threadIdx.x >> 5;
  const int e = blockIdx.x*4 + wid;
  char* wb = smem + wid*K1_WAVE;
  float* feats = (float*)wb;           // 264 f32
  f16*   x     = (f16*)(wb + 1056);    // 32x128
  f16*   h     = (f16*)(wb + 9248);    // 32x128
  float* es    = (float*)(wb + 17440); // 68
  float* ed    = (float*)(wb + 17712); // 68

  const float* st = p.state + e*4;
  const float* ch = p.chase + e*3;
  const float* hr = p.hr + e*24;
  const float* hw = p.hw + e*24;

  float sum_hr=0, balls=0, dots=0, bnd=0, wkts=0, rec_sum=0, p_balls=0, consec=0;
  for (int s = 0; s < 24; ++s) {
    float r = hr[s];
    sum_hr += r;
    balls += (r > -1.f) ? 1.f : 0.f;
    dots  += (r == 0.f) ? 1.f : 0.f;
    bnd   += (r >= 4.f/6.f) ? 1.f : 0.f;
    wkts  += hw[s];
    if (s >= 12) { rec_sum += r; p_balls += (r > -1.f) ? 1.f : 0.f; }
    if (s >= 18) consec += (r == 0.f) ? 1.f : 0.f;
  }
  float last_bnd = 0.f;
  for (int s = 23; s >= 0; --s) { if (hr[s] >= 4.f/6.f) break; last_bnd += 1.f; }

  // gathered embeddings
  feats[lane]      = p.venue_emb[p.venue_idx[e]*32 + lane];
  feats[32 + lane] = p.team_emb[p.batting_team_idx[e]*32 + lane];
  feats[64 + lane] = p.team_emb[p.bowling_team_idx[e]*32 + lane];
  {
    int bi = p.batter_idx[e], bo = p.bowler_idx[e];
    feats[112 + lane] = p.player_emb[bi*64 + lane];
    feats[144 + lane] = p.player_emb[bi*64 + 32 + lane];
    feats[182 + lane] = p.player_emb[bo*64 + lane];
    feats[214 + lane] = p.player_emb[bo*64 + 32 + lane];
  }
  if (lane == 0) {
    float op = st[2], wick = st[1]*10.f;
    feats[96]=st[0]; feats[97]=st[1]; feats[98]=st[2]; feats[99]=st[3];
    feats[100]=ch[0]; feats[101]=ch[1]; feats[102]=ch[2];
    float over = op*20.f;
    feats[103] = (over < 6.f) ? 1.f : 0.f;
    feats[104] = (over >= 6.f && over < 15.f) ? 1.f : 0.f;
    feats[105] = (over >= 15.f) ? 1.f : 0.f;
    feats[106] = op;
    float br = 1.f - op;
    feats[107]=br; feats[108]=1.f-br; feats[109]=(br < 0.25f) ? 1.f : 0.f;
    feats[110]=1.f - wick/10.f; feats[111]=(wick > 0.7f) ? 1.f : 0.f;
    float runs = sum_hr*6.f, bc = fmaxf(balls, 1.f), sr = runs/bc*100.f;
    feats[176]=runs/100.f; feats[177]=balls/60.f; feats[178]=sr/200.f;
    feats[179]=dots/bc; feats[180]=fminf(balls/30.f,1.f); feats[181]=bnd/10.f;
    float econ = runs / fmaxf(balls/6.f, 0.1f);
    float threat = 0.5f*(1.f-econ/12.f) + 0.3f*(wkts/4.f) + 0.2f*(dots/bc);
    feats[246]=balls/24.f; feats[247]=runs/50.f; feats[248]=wkts/4.f;
    feats[249]=econ/12.f; feats[250]=dots/bc; feats[251]=threat;
    float p_runs = rec_sum*6.f;
    float p_rr = p_runs / fmaxf(p_balls, 0.1f) * 6.f;
    feats[252]=p_runs/100.f; feats[253]=p_balls/60.f; feats[254]=p_rr/12.f;
    feats[255]=fminf(p_balls/30.f,1.f);
    float mom = fminf(fmaxf(rec_sum*6.f/48.f*2.f - 1.f, -1.f), 1.f);
    feats[256]=mom; feats[257]=-mom;
    feats[258]=fminf(st[1]*0.3f + ch[2]*fmaxf(ch[1],0.f)*0.4f + ((op > 0.75f)?0.1f:0.f), 1.f);
    feats[259]=consec/6.f; feats[260]=last_bnd/30.f;
  }
  for (int idx = lane; idx < 15*128; idx += 32) x[17*128 + idx] = (f16)0.f;
  __syncthreads();

  // ragged projection (261 -> 17x128), VALU
  for (int idx = lane; idx < 17*128; idx += 32) {
    int n = idx >> 7, c = idx & 127;
    int si = PROJ_IDX[n];
    const float* w = p.proj_w[si];
    float acc = p.proj_b[si][c];
    int off = FOFF[n], d = FDIM[n];
    for (int j = 0; j < d; ++j) acc += feats[off + j] * w[j*128 + c];
    x[idx] = (f16)acc;
  }
  __syncthreads();

#pragma unroll 1
  for (int l = 0; l < 4; ++l) {
    const f16* W = p.wsW + GATW_BASE + l*16384;
    for (int rt = 0; rt < 2; ++rt)
      gemm_a_res<4,8>(x, 128, W, lane, rt, [&](int nt, v8f acc){
        int n  = nt*16 + (lane & 15);
        int mb = rt*16 + ((lane >> 4) << 3);
#pragma unroll
        for (int v = 0; v < 8; ++v) h[(mb + v)*128 + n] = (f16)acc[v];
      });
    __syncthreads();
    const float* asrc = p.gat_a_src[l];
    const float* adst = p.gat_a_dst[l];
    for (int t = lane; t < 68; t += 32) {
      int head = t / 17, n = t - head*17;
      float se = 0.f, de = 0.f;
      for (int d = 0; d < 32; ++d) {
        float hv = (float)h[n*128 + head*32 + d];
        se += hv * asrc[head*32 + d];
        de += hv * adst[head*32 + d];
      }
      es[head*17 + n] = se; ed[head*17 + n] = de;
    }
    __syncthreads();
    for (int t = lane; t < 68; t += 32) {
      int head = t & 3, i = t >> 2;
      float si_ = es[head*17 + i];
      float sc[17]; float mx = -1e30f;
#pragma unroll
      for (int j = 0; j < 17; ++j) {
        float v2 = si_ + ed[head*17 + j];
        v2 = (v2 > 0.f) ? v2 : 0.2f*v2;
        sc[j] = v2; mx = fmaxf(mx, v2);
      }
      float sm = 0.f;
#pragma unroll
      for (int j = 0; j < 17; ++j) { sc[j] = __expf(sc[j] - mx); sm += sc[j]; }
      float inv = 1.f / sm;
      for (int d = 0; d < 32; ++d) {
        float o = 0.f;
#pragma unroll
        for (int j = 0; j < 17; ++j) o += sc[j] * (float)h[j*128 + head*32 + d];
        o *= inv;
        float r = o + (float)x[i*128 + head*32 + d];
        r = (r > 0.f) ? r : (__expf(r) - 1.f);
        x[i*128 + head*32 + d] = (f16)r;
      }
    }
    __syncthreads();
  }
  for (int c = lane; c < 128; c += 32) {
    float s = 0.f;
    for (int n = 0; n < 17; ++n) s += (float)x[n*128 + c];
    p.gat_out[e*128 + c] = s * (1.f/17.f);
  }
}

// -------------------- kernel 2: token init + transformer --------------------
#define K2_WAVE 57344
__device__ __forceinline__ void layernorm24(const f16* src, f16* dst,
                                            const float* s_, const float* b_, int lane) {
  if (lane < 24) {
    const f16* row = src + lane*128;
    float s = 0.f, s2 = 0.f;
    for (int c = 0; c < 128; ++c) { float v = (float)row[c]; s += v; s2 += v*v; }
    float m   = s * (1.f/128.f);
    float var = s2 * (1.f/128.f) - m*m;
    float inv = rsqrtf(var + 1e-5f);
    for (int c = 0; c < 128; ++c)
      dst[lane*128 + c] = (f16)(((float)row[c] - m)*inv*s_[c] + b_[c]);
  }
}

__global__ void __launch_bounds__(128) k_trans(P p) {
  extern __shared__ char smem[];
  const int lane = threadIdx.x & 31;
  const int wid  = threadIdx.x >> 5;
  const int e = blockIdx.x*4 + wid;
  char* wb = smem + wid*K2_WAVE;
  f16* tok = (f16*)wb;             // 32x128
  f16* tn  = (f16*)(wb + 8192);    // 32x128
  f16* att = (f16*)(wb + 16384);   // 32x128
  f16* big = (f16*)(wb + 24576);   // up to 32x512

  const float* hr = p.hr + e*24;
  const float* hw = p.hw + e*24;
  const float* ho = p.ho + e*24;
  const int*   hb  = p.hb  + e*24;
  const int*   hbl = p.hbl + e*24;

  for (int idx = lane; idx < 8*128; idx += 32) {
    tn[24*128 + idx]  = (f16)0.f;
    att[24*128 + idx] = (f16)0.f;
  }
  // build combined token-init A (32x160): [hr,hw,ho | pe[hb] | pe[hbl] | 0-pad]
  for (int idx = lane; idx < 32*160; idx += 32) {
    int r = idx / 160, c = idx - r*160;
    float v = 0.f;
    if (r < 24) {
      if (c < 3)        v = (c == 0) ? hr[r] : (c == 1) ? hw[r] : ho[r];
      else if (c < 67)  v = p.player_emb[hb[r]*64 + (c - 3)];
      else if (c < 131) v = p.player_emb[hbl[r]*64 + (c - 67)];
    }
    big[idx] = (f16)v;
  }
  __syncthreads();

  for (int rt = 0; rt < 2; ++rt)
    gemm_a_res<5,8>(big, 160, p.wsW + TOKINIT_BASE, lane, rt, [&](int nt, v8f acc){
      int n  = nt*16 + (lane & 15);
      int mb = rt*16 + ((lane >> 4) << 3);
#pragma unroll
      for (int v = 0; v < 8; ++v) {
        int m = mb + v;
        float val = (m < 24) ? (acc[v] + p.t_feat_b[n] + p.pos[m*128 + n]) : 0.f;
        tok[m*128 + n] = (f16)val;
      }
    });
  __syncthreads();

#pragma unroll 1
  for (int l = 0; l < 2; ++l) {
    layernorm24(tok, tn, p.tl_ln1s[l], p.tl_ln1b[l], lane);
    __syncthreads();
    // qkv = tn @ wqkv   (32x384 into big)
    for (int rt = 0; rt < 2; ++rt)
      gemm_a_res<4,24>(tn, 128, p.wsW + WQKV_BASE + l*TL_STRIDE, lane, rt, [&](int nt, v8f acc){
        int n  = nt*16 + (lane & 15);
        int mb = rt*16 + ((lane >> 4) << 3);
#pragma unroll
        for (int v = 0; v < 8; ++v) { int m = mb + v; big[m*384 + n] = (f16)((m < 24) ? acc[v] : 0.f); }
      });
    __syncthreads();
    // attention (4 heads, dh=32) in VALU
    for (int t = lane; t < 96; t += 32) {
      int head = t & 3, i = t >> 2;
      const f16* q = big + i*384 + head*32;
      float sc[24]; float mx = -1e30f;
      for (int j = 0; j < 24; ++j) {
        const f16* kk = big + j*384 + 128 + head*32;
        float s = 0.f;
#pragma unroll
        for (int d = 0; d < 32; ++d) s += (float)q[d] * (float)kk[d];
        s *= 0.17677669529663687f;  // 1/sqrt(32)
        sc[j] = s; mx = fmaxf(mx, s);
      }
      float sm = 0.f;
#pragma unroll
      for (int j = 0; j < 24; ++j) { sc[j] = __expf(sc[j] - mx); sm += sc[j]; }
      float inv = 1.f / sm;
      for (int d = 0; d < 32; ++d) {
        float o = 0.f;
#pragma unroll
        for (int j = 0; j < 24; ++j) o += sc[j] * (float)big[j*384 + 256 + head*32 + d];
        att[i*128 + head*32 + d] = (f16)(o * inv);
      }
    }
    __syncthreads();
    // out-proj + residual
    for (int rt = 0; rt < 2; ++rt)
      gemm_a_res<4,8>(att, 128, p.wsW + WO_BASE + l*TL_STRIDE, lane, rt, [&](int nt, v8f acc){
        int n  = nt*16 + (lane & 15);
        int mb = rt*16 + ((lane >> 4) << 3);
#pragma unroll
        for (int v = 0; v < 8; ++v) {
          int m = mb + v;
          float val = (m < 24) ? (acc[v] + (float)tok[m*128 + n]) : 0.f;
          tok[m*128 + n] = (f16)val;
        }
      });
    __syncthreads();
    layernorm24(tok, tn, p.tl_ln2s[l], p.tl_ln2b[l], lane);
    __syncthreads();
    // FFN up: relu(tn @ w1 + b1) -> big (32x512)
    for (int rt = 0; rt < 2; ++rt)
      gemm_a_res<4,32>(tn, 128, p.wsW + W1_BASE + l*TL_STRIDE, lane, rt, [&](int nt, v8f acc){
        int n  = nt*16 + (lane & 15);
        int mb = rt*16 + ((lane >> 4) << 3);
        const float* b1 = p.tl_b1[l];
#pragma unroll
        for (int v = 0; v < 8; ++v) {
          int m = mb + v;
          float val = (m < 24) ? fmaxf(acc[v] + b1[n], 0.f) : 0.f;
          big[m*512 + n] = (f16)val;
        }
      });
    __syncthreads();
    // FFN down + residual (K=512: accumulator-resident)
    for (int rt = 0; rt < 2; ++rt)
      gemm_c_res<16,8>(big, 512, p.wsW + W2_BASE + l*TL_STRIDE, lane, rt, [&](int nt, v8f acc){
        int n  = nt*16 + (lane & 15);
        int mb = rt*16 + ((lane >> 4) << 3);
        const float* b2 = p.tl_b2[l];
#pragma unroll
        for (int v = 0; v < 8; ++v) {
          int m = mb + v;
          float val = (m < 24) ? ((float)tok[m*128 + n] + acc[v] + b2[n]) : 0.f;
          tok[m*128 + n] = (f16)val;
        }
      });
    __syncthreads();
  }
  for (int c = lane; c < 128; c += 32)
    p.temporal_out[e*128 + c] = (float)tok[23*128 + c];
}

// -------------------- kernel 3: fuse MLP + output --------------------
#define K3_WAVE 14336
__global__ void __launch_bounds__(128) k_fuse(P p, float* out) {
  extern __shared__ char smem[];
  const int lane = threadIdx.x & 31;
  const int wid  = threadIdx.x >> 5;
  const int e0 = (blockIdx.x*4 + wid) * 16;
  char* wb = smem + wid*K3_WAVE;
  f16* af = (f16*)wb;             // 16x256
  f16* h1 = (f16*)(wb + 8192);    // 16x128
  f16* h2 = (f16*)(wb + 12288);   // 16x64
  for (int idx = lane; idx < 16*256; idx += 32) {
    int r = idx >> 8, c = idx & 255;
    float v = (c < 128) ? p.gat_out[(e0 + r)*128 + c]
                        : p.temporal_out[(e0 + r)*128 + (c - 128)];
    af[idx] = (f16)v;
  }
  __syncthreads();
  gemm_a_res<8,8>(af, 256, p.wsW + FUSE1_BASE, lane, 0, [&](int nt, v8f acc){
    int n  = nt*16 + (lane & 15);
    int mb = (lane >> 4) << 3;
#pragma unroll
    for (int v = 0; v < 8; ++v) h1[(mb + v)*128 + n] = (f16)fmaxf(acc[v] + p.fuse_b1[n], 0.f);
  });
  __syncthreads();
  gemm_a_res<4,4>(h1, 128, p.wsW + FUSE2_BASE, lane, 0, [&](int nt, v8f acc){
    int n  = nt*16 + (lane & 15);
    int mb = (lane >> 4) << 3;
#pragma unroll
    for (int v = 0; v < 8; ++v) h2[(mb + v)*64 + n] = (f16)fmaxf(acc[v] + p.fuse_b2[n], 0.f);
  });
  __syncthreads();
  gemm_a_res<2,1>(h2, 64, p.wsW + OUTW_BASE, lane, 0, [&](int nt, v8f acc){
    (void)nt;
    int n  = lane & 15;
    int mb = (lane >> 4) << 3;
    if (n < 7) {
#pragma unroll
      for (int v = 0; v < 8; ++v) out[(e0 + mb + v)*7 + n] = acc[v] + p.out_b[n];
    }
  });
}

// -------------------- host launcher --------------------
extern "C" void kernel_launch(void* const* d_in, const int* in_sizes, int n_in,
                              void* d_out, int out_size, void* d_ws, size_t ws_size,
                              hipStream_t stream) {
  (void)in_sizes; (void)n_in; (void)out_size; (void)ws_size;
  P p;
  auto F = [&](int i){ return (const float*)d_in[i]; };
  auto I = [&](int i){ return (const int*)d_in[i]; };
  // params leaves in jax tree-flatten (sorted-key) order
  p.fuse_b1 = F(0); p.fuse_b2 = F(1); p.fuse_w1 = F(2); p.fuse_w2 = F(3);
  for (int l = 0; l < 4; ++l) {
    p.gat_a_dst[l] = F(4 + 3*l);
    p.gat_a_src[l] = F(5 + 3*l);
    p.gat_w[l]     = F(6 + 3*l);
  }
  p.out_b = F(16); p.out_w = F(17); p.player_emb = F(18); p.pos = F(19);
  for (int i = 0; i < 17; ++i) { p.proj_b[i] = F(20 + 2*i); p.proj_w[i] = F(21 + 2*i); }
  p.t_bat_w = F(54); p.t_bowl_w = F(55); p.t_feat_b = F(56); p.t_feat_w = F(57);
  p.team_emb = F(58);
  for (int l = 0; l < 2; ++l) {
    int b = 59 + 10*l;
    p.tl_b1[l]=F(b); p.tl_b2[l]=F(b+1); p.tl_ln1b[l]=F(b+2); p.tl_ln1s[l]=F(b+3);
    p.tl_ln2b[l]=F(b+4); p.tl_ln2s[l]=F(b+5); p.tl_w1[l]=F(b+6); p.tl_w2[l]=F(b+7);
    p.tl_wo[l]=F(b+8); p.tl_wqkv[l]=F(b+9);
  }
  p.venue_emb = F(79);
  p.batter_idx = I(80); p.bowler_idx = I(81); /* 82: non_striker unused */
  p.venue_idx = I(83); p.batting_team_idx = I(84); p.bowling_team_idx = I(85);
  p.state = F(86); p.chase = F(87);
  p.hr = F(88); p.hw = F(89); p.ho = F(90);
  p.hb = I(91); p.hbl = I(92);

  p.wsW          = (f16*)d_ws;
  p.gat_out      = (float*)((char*)d_ws + (1u  << 20));
  p.temporal_out = (float*)((char*)d_ws + (9u  << 20));

  k_prep <<<1018, 32, 0, stream>>>(p);
  k_gat  <<<B_TOTAL/4, 128, 4*K1_WAVE, stream>>>(p);
  k_trans<<<B_TOTAL/4, 128, 4*K2_WAVE, stream>>>(p);
  k_fuse <<<B_TOTAL/64, 128, 4*K3_WAVE, stream>>>(p, (float*)d_out);
}
